// GATModel2_28089086116669
// MI455X (gfx1250) — compile-verified, matrix operand order checked by reading, not run on hardware
//
#include <hip/hip_runtime.h>
#include <hip/hip_bf16.h>

// Problem constants (match reference)
#define E_EDGES 8192
#define C_DIM   128
#define RTILES  (E_EDGES / 16)   // 512 row tiles of 16
#define CHUNKS  (E_EDGES / 32)   // 256 source chunks of 32
#define NTILES  (C_DIM / 16)     // 8 column tiles of 16
#define KCHUNKS (C_DIM / 32)     // 4 K chunks of 32 for the projection GEMM

typedef __attribute__((ext_vector_type(16))) __bf16 v16bf;
typedef __attribute__((ext_vector_type(8)))  float  v8f;

// float -> bf16 round-to-nearest-even
__device__ __forceinline__ __bf16 f2bf(float f) {
  unsigned u = __builtin_bit_cast(unsigned, f);
  unsigned r = u + 0x7FFFu + ((u >> 16) & 1u);
  unsigned short h = (unsigned short)(r >> 16);
  return __builtin_bit_cast(__bf16, h);
}

// CDNA5 16-bit WMMA operand K mapping (ISA 7.12.2):
// lanes 0-15 hold K in {0..7, 16..23}; lanes 16-31 hold K in {8..15, 24..31},
// element e packs as (e<8 ? e : e+8) within the half's base.
__device__ __forceinline__ int kmap(int e, int l) {
  return ((l >= 16) ? 8 : 0) + ((e < 8) ? e : e + 8);
}

// --- Kernel 1: x [E,128] fp32 -> A-layout bf16 tiles ------------------------
__global__ void prep_x_kernel(const float* __restrict__ x, __bf16* __restrict__ xA) {
  int idx = blockIdx.x * blockDim.x + threadIdx.x;     // E*C elements
  if (idx >= E_EDGES * C_DIM) return;
  int e  = idx & 15;
  int l  = (idx >> 4) & 31;
  int kc = (idx >> 9) & 3;
  int rt = idx >> 11;
  int row = rt * 16 + (l & 15);
  int col = kc * 32 + kmap(e, l);
  xA[idx] = f2bf(x[row * C_DIM + col]);
}

// --- Kernel 2: W [128,128] fp32 -> B-layout bf16 tiles ----------------------
__global__ void prep_W_kernel(const float* __restrict__ W, __bf16* __restrict__ wB) {
  int idx = blockIdx.x * blockDim.x + threadIdx.x;     // 4*8*32*16 elements
  if (idx >= KCHUNKS * NTILES * 32 * 16) return;
  int e  = idx & 15;
  int l  = (idx >> 4) & 31;
  int nt = (idx >> 9) & 7;
  int kc = idx >> 12;
  int row = kc * 32 + kmap(e, l);
  int col = nt * 16 + (l & 15);
  wB[idx] = f2bf(W[row * C_DIM + col]);
}

// --- Kernel 3: h = x @ W via v_wmma_f32_16x16x32_bf16 -----------------------
__global__ void gemm_h_kernel(const __bf16* __restrict__ xA,
                              const __bf16* __restrict__ wB,
                              float* __restrict__ h) {
  int wave = (blockIdx.x * blockDim.x + threadIdx.x) >> 5;  // row tile rt
  int l = threadIdx.x & 31;
  v8f acc[NTILES];
#pragma unroll
  for (int nt = 0; nt < NTILES; ++nt) acc[nt] = (v8f)0.0f;

#pragma unroll
  for (int kc = 0; kc < KCHUNKS; ++kc) {
    v16bf a = *(const v16bf*)(xA + ((size_t)(wave * KCHUNKS + kc) * 32 + l) * 16);
#pragma unroll
    for (int nt = 0; nt < NTILES; ++nt) {
      v16bf b = *(const v16bf*)(wB + ((size_t)(kc * NTILES + nt) * 32 + l) * 16);
      acc[nt] = __builtin_amdgcn_wmma_f32_16x16x32_bf16(
          false, a, false, b, (short)0, acc[nt], false, false);
    }
  }
  // C/D layout: VGPR v -> M = v (+8 for lanes 16-31), N = lane%16
  int hl8 = (l >= 16) ? 8 : 0;
  int cb  = l & 15;
#pragma unroll
  for (int nt = 0; nt < NTILES; ++nt)
#pragma unroll
    for (int v = 0; v < 8; ++v)
      h[(size_t)(wave * 16 + v + hl8) * C_DIM + nt * 16 + cb] = acc[nt][v];
}

// --- Kernel 4: attention logits a_src = h.att_src, a_dst = h.att_dst --------
__global__ void logits_kernel(const float* __restrict__ h,
                              const float* __restrict__ att_src,
                              const float* __restrict__ att_dst,
                              float* __restrict__ a_src,
                              float* __restrict__ a_dst) {
  int t = blockIdx.x * blockDim.x + threadIdx.x;
  int row = t >> 5;
  int lane = t & 31;
  if (row >= E_EDGES) return;
  float ss = 0.f, dd = 0.f;
#pragma unroll
  for (int c = lane; c < C_DIM; c += 32) {
    float hv = h[(size_t)row * C_DIM + c];
    ss += hv * att_src[c];
    dd += hv * att_dst[c];
  }
#pragma unroll
  for (int o = 16; o > 0; o >>= 1) {
    ss += __shfl_xor(ss, o, 32);
    dd += __shfl_xor(dd, o, 32);
  }
  if (lane == 0) { a_src[row] = ss; a_dst[row] = dd; }
}

// --- Kernel 5: h fp32 -> B-layout bf16 (value matrix for attention) ---------
__global__ void reorder_h_kernel(const float* __restrict__ h, __bf16* __restrict__ hB) {
  int idx = blockIdx.x * blockDim.x + threadIdx.x;     // E*C elements
  if (idx >= E_EDGES * C_DIM) return;
  int e  = idx & 15;
  int l  = (idx >> 4) & 31;
  int nt = (idx >> 9) & 7;
  int c  = idx >> 12;
  int j = c * 32 + kmap(e, l);
  int n = nt * 16 + (l & 15);
  hB[idx] = f2bf(h[(size_t)j * C_DIM + n]);
}

// --- Kernel 6: flash-style masked attention over the line graph -------------
// One wave per 16 target rows; streams 256 chunks of 32 source edges.
__global__ void flash_kernel(const int* __restrict__ s, const int* __restrict__ d,
                             const float* __restrict__ a_src,
                             const float* __restrict__ a_dst,
                             const __bf16* __restrict__ hB,
                             const float* __restrict__ bias,
                             float* __restrict__ out) {
  int wave = (blockIdx.x * blockDim.x + threadIdx.x) >> 5;  // target tile
  int l = threadIdx.x & 31;
  int r = l & 15;
  int i0 = wave * 16;
  int i  = i0 + r;                      // this lane's target row
  int si = s[i], di = d[i];
  float adsti = a_dst[i];

  int kofs[16];
#pragma unroll
  for (int e = 0; e < 16; ++e) kofs[e] = kmap(e, l);

  v8f acc[NTILES];
#pragma unroll
  for (int nt = 0; nt < NTILES; ++nt) acc[nt] = (v8f)0.0f;
  float mrun = -1e30f, lrun = 0.f;
  int hl8 = (l >= 16) ? 8 : 0;

  for (int c = 0; c < CHUNKS; ++c) {
    int j0 = c * 32;
    // each lane grabs one source edge, then shuffles per K element
    int   sj  = s[j0 + l];
    int   dj  = d[j0 + l];
    float asj = a_src[j0 + l];

    float sc[16];
    float mloc = -1e30f;
#pragma unroll
    for (int e = 0; e < 16; ++e) {
      int kk = kofs[e];
      int   sjk = __shfl(sj, kk, 32);
      int   djk = __shfl(dj, kk, 32);
      float ajk = __shfl(asj, kk, 32);
      int j = j0 + kk;
      bool adj = (i == j) | (si == sjk) | (si == djk) | (di == sjk) | (di == djk);
      float v = adsti + ajk;
      v = (v > 0.f) ? v : 0.2f * v;               // leaky relu
      sc[e] = adj ? v : -1e30f;
      mloc = fmaxf(mloc, sc[e]);
    }
    // row max across the two K halves (lanes r and r+16 share row r)
    mloc = fmaxf(mloc, __shfl_xor(mloc, 16, 32));
    float mnew = fmaxf(mrun, mloc);
    float cs = __expf(mrun - mnew);               // rescale factor

    float lloc = 0.f;
    v16bf a;
#pragma unroll
    for (int e = 0; e < 16; ++e) {
      float p = __expf(sc[e] - mnew);
      lloc += p;
      a[e] = f2bf(p);
    }
    lloc += __shfl_xor(lloc, 16, 32);
    lrun = lrun * cs + lloc;
    mrun = mnew;

    // broadcast per-row rescale factor (row m lives in lane m)
    float cm[16];
#pragma unroll
    for (int m = 0; m < 16; ++m) cm[m] = __shfl(cs, m, 32);

#pragma unroll
    for (int nt = 0; nt < NTILES; ++nt) {
      v16bf b = *(const v16bf*)(hB + ((size_t)(c * NTILES + nt) * 32 + l) * 16);
#pragma unroll
      for (int v = 0; v < 8; ++v) acc[nt][v] *= cm[v + hl8];
      acc[nt] = __builtin_amdgcn_wmma_f32_16x16x32_bf16(
          false, a, false, b, (short)0, acc[nt], false, false);
    }
  }

  // epilogue: divide by row sum, add bias, store
  float lm[16];
#pragma unroll
  for (int m = 0; m < 16; ++m) lm[m] = __shfl(lrun, m, 32);
#pragma unroll
  for (int nt = 0; nt < NTILES; ++nt) {
    float bcol = bias[nt * 16 + r];
#pragma unroll
    for (int v = 0; v < 8; ++v) {
      int m = v + hl8;
      out[(size_t)(i0 + m) * C_DIM + nt * 16 + r] = acc[nt][v] / lm[m] + bcol;
    }
  }
}

extern "C" void kernel_launch(void* const* d_in, const int* in_sizes, int n_in,
                              void* d_out, int out_size, void* d_ws, size_t ws_size,
                              hipStream_t stream) {
  const float* x       = (const float*)d_in[0];
  const int*   ei      = (const int*)d_in[1];   // edge_index [2,E]
  const float* W       = (const float*)d_in[2];
  const float* att_src = (const float*)d_in[3];
  const float* att_dst = (const float*)d_in[4];
  const float* bias    = (const float*)d_in[5];
  float*       out     = (float*)d_out;

  char* ws = (char*)d_ws;
  __bf16* xA    = (__bf16*)(ws + 0);                         // 2 MB
  float*  h     = (float*) (ws + 2u * 1024 * 1024);          // 4 MB
  __bf16* hB    = (__bf16*)(ws + 6u * 1024 * 1024);          // 2 MB
  __bf16* wB    = (__bf16*)(ws + 8u * 1024 * 1024);          // 32 KB
  float*  asrc  = (float*) (ws + 8u * 1024 * 1024 + 64u * 1024);
  float*  adst  = asrc + E_EDGES;

  const int* s = ei;
  const int* d = ei + E_EDGES;

  prep_x_kernel   <<<(E_EDGES * C_DIM) / 256, 256, 0, stream>>>(x, xA);
  prep_W_kernel   <<<(KCHUNKS * NTILES * 32 * 16) / 256, 256, 0, stream>>>(W, wB);
  gemm_h_kernel   <<<RTILES / 8, 256, 0, stream>>>(xA, wB, h);
  logits_kernel   <<<(E_EDGES * 32) / 256, 256, 0, stream>>>(h, att_src, att_dst, asrc, adst);
  reorder_h_kernel<<<(E_EDGES * C_DIM) / 256, 256, 0, stream>>>(h, hB);
  flash_kernel    <<<RTILES / 8, 256, 0, stream>>>(s, d, asrc, adst, hB, bias, out);
}
